// HungarianMatcher_21337397526859
// MI455X (gfx1250) — matrix operand order repaired and checked by reading, not compiled, not run on hardware
//
#include <hip/hip_runtime.h>
#include <math.h>

// ---------------------------------------------------------------------------
// HungarianMatcher cost matrix for MI455X (gfx1250, wave32, WMMA).
// Cmat[b,q,t] = 5*cost_mask + 2*cost_class + 5*cost_dice
// Key identity: softplus(-x) - softplus(x) = -x  =>
//   cost_mask*P = sum_p softplus(om) - sum_p om*tm      (one GEMM instead of two)
//   cost_dice needs sum_p sigmoid(om)*tm                (second GEMM)
// GEMMs are M=100,N=50,K=12544 per batch -> v_wmma_f32_16x16x32_f16 tiles.
// ---------------------------------------------------------------------------

typedef __attribute__((ext_vector_type(16))) _Float16 v16h;
typedef __attribute__((ext_vector_type(8)))  _Float16 h8;
typedef __attribute__((ext_vector_type(8)))  float    v8f;

namespace {
constexpr int kB = 8, kQ = 100, kC = 41, kT = 50, kH = 256, kW = 256, kP = 12544;
constexpr int kChunks = kP / 256;            // 49 blocks of 256 points per row
constexpr int kQTiles = (kQ + 15) / 16;      // 7
constexpr int kTTiles = (kT + 15) / 16;      // 4
}

// ---------------- bilinear point sample (grid_sample, zeros padding) -------
__device__ __forceinline__ float tap(const float* __restrict__ m, int x, int y) {
  if (x < 0 || x >= kW || y < 0 || y >= kH) return 0.0f;
  return m[y * kW + x];
}

__device__ __forceinline__ float point_sample(const float* __restrict__ m,
                                              float cx, float cy) {
  float x = cx * (float)kW - 0.5f;
  float y = cy * (float)kH - 0.5f;
  float x0f = floorf(x), y0f = floorf(y);
  float wx = x - x0f, wy = y - y0f;
  int x0 = (int)x0f, y0 = (int)y0f;
  float v00 = tap(m, x0,     y0);
  float v01 = tap(m, x0 + 1, y0);
  float v10 = tap(m, x0,     y0 + 1);
  float v11 = tap(m, x0 + 1, y0 + 1);
  return v00 * (1.0f - wx) * (1.0f - wy) + v01 * wx * (1.0f - wy) +
         v10 * (1.0f - wx) * wy          + v11 * wx * wy;
}

// ---------------- phase 1: sample target masks -> f16 + partial row sums ---
__global__ void sample_tgt_kernel(const float* __restrict__ tgt_masks,
                                  const float* __restrict__ coords,
                                  _Float16* __restrict__ tm16,
                                  float* __restrict__ partTm) {
  const int tid = threadIdx.x;
  const int p = blockIdx.x * 256 + tid;
  const int t = blockIdx.y, b = blockIdx.z;
  const float cx = coords[((size_t)b * kP + p) * 2 + 0];
  const float cy = coords[((size_t)b * kP + p) * 2 + 1];
  const float* m = tgt_masks + (size_t)(b * kT + t) * (kH * kW);
  const float v = point_sample(m, cx, cy);
  tm16[(size_t)(b * kT + t) * kP + p] = (_Float16)v;

  __shared__ float red[256];
  red[tid] = v;
  __syncthreads();
  for (int s = 128; s > 0; s >>= 1) {
    if (tid < s) red[tid] += red[tid + s];
    __syncthreads();
  }
  if (tid == 0) partTm[(b * kT + t) * kChunks + blockIdx.x] = red[0];
}

// ---------------- phase 2: sample pred masks -> om/sig f16 + partial sums --
__global__ void sample_pred_kernel(const float* __restrict__ pred_masks,
                                   const float* __restrict__ coords,
                                   _Float16* __restrict__ om16,
                                   _Float16* __restrict__ sg16,
                                   float* __restrict__ partNeg,
                                   float* __restrict__ partSig) {
  const int tid = threadIdx.x;
  const int p = blockIdx.x * 256 + tid;
  const int q = blockIdx.y, b = blockIdx.z;
  const float cx = coords[((size_t)b * kP + p) * 2 + 0];
  const float cy = coords[((size_t)b * kP + p) * 2 + 1];
  const float* m = pred_masks + (size_t)(b * kQ + q) * (kH * kW);
  const float v = point_sample(m, cx, cy);
  const float sig = 1.0f / (1.0f + expf(-v));
  // numerically stable softplus(v)
  const float sp = fmaxf(v, 0.0f) + log1pf(expf(-fabsf(v)));
  const size_t o = (size_t)(b * kQ + q) * kP + p;
  om16[o] = (_Float16)v;
  sg16[o] = (_Float16)sig;

  __shared__ float redA[256];
  __shared__ float redB[256];
  redA[tid] = sp;
  redB[tid] = sig;
  __syncthreads();
  for (int s = 128; s > 0; s >>= 1) {
    if (tid < s) { redA[tid] += redA[tid + s]; redB[tid] += redB[tid + s]; }
    __syncthreads();
  }
  if (tid == 0) {
    partNeg[(b * kQ + q) * kChunks + blockIdx.x] = redA[0];
    partSig[(b * kQ + q) * kChunks + blockIdx.x] = redB[0];
  }
}

// ---------------- phase 3: deterministic reduction of 49 partials ----------
__global__ void reduce_parts_kernel(const float* __restrict__ part,
                                    float* __restrict__ out, int n) {
  const int i = blockIdx.x * blockDim.x + threadIdx.x;
  if (i >= n) return;
  float s = 0.0f;
  for (int j = 0; j < kChunks; ++j) s += part[(size_t)i * kChunks + j];
  out[i] = s;
}

// ---------------- WMMA operand packing (ISA §7.12.2 16-bit layouts) --------
// A 16x32 f16: lanes 0-15 = rows M, halves = K {0..7,16..23}; lanes 16-31 same
// rows, K {8..15,24..31}. Both chunks are 16B-contiguous in row-major memory.
__device__ __forceinline__ v16h pack_a(const _Float16* __restrict__ p) {
  h8 lo = *(const h8*)(p);
  h8 hh = *(const h8*)(p + 16);
  v16h r;
#pragma unroll
  for (int i = 0; i < 8; ++i) { r[i] = lo[i]; r[i + 8] = hh[i]; }
  return r;
}
// B 32x16 f16: lanes 0-15 = cols N, K {0..7,16..23}; lanes 16-31 same
// cols, K {8..15,24..31} (same K order as A).
__device__ __forceinline__ v16h pack_b(const _Float16* __restrict__ p) {
  h8 lo = *(const h8*)(p);
  h8 hh = *(const h8*)(p + 16);
  v16h r;
#pragma unroll
  for (int i = 0; i < 8; ++i) { r[i] = lo[i]; r[i + 8] = hh[i]; }
  return r;
}

// ---------------- phase 4: WMMA GEMMs + fused cost epilogue ----------------
__global__ void __launch_bounds__(32)
cost_matmul_kernel(const _Float16* __restrict__ om16,
                   const _Float16* __restrict__ sg16,
                   const _Float16* __restrict__ tm16,
                   const float* __restrict__ negsum,   // [B*Q] sum softplus(om)
                   const float* __restrict__ sigsum,   // [B*Q] sum sigmoid(om)
                   const float* __restrict__ tmsum,    // [B*T] sum tm
                   const float* __restrict__ plog,     // pred_logits [B,Q,C]
                   const float* __restrict__ pobj,     // pred_object_logits [B,Q,2]
                   const int*   __restrict__ labels,   // tgt_labels [B,T]
                   float* __restrict__ out) {          // [B,Q,T]
  const int lane = threadIdx.x;         // one wave32 per 16x16 output tile
  const int ll = lane & 15;
  const int hi = lane >> 4;
  const int qBase = blockIdx.x * 16;
  const int tBase = blockIdx.y * 16;
  const int b = blockIdx.z;

  // Clamp pad rows/cols so EXEC stays all-ones through the WMMA loop
  // (ISA: EXEC must be all 1s for WMMA); pad results are discarded below.
  const int qr = min(qBase + ll, kQ - 1);
  const int tr = min(tBase + ll, kT - 1);
  const _Float16* aRow = om16 + (size_t)(b * kQ + qr) * kP;
  const _Float16* sRow = sg16 + (size_t)(b * kQ + qr) * kP;
  const _Float16* bRow = tm16 + (size_t)(b * kT + tr) * kP;

  const int ka = hi * 8;    // A per-lane-half K offset within a 32-block
  const int kb = hi * 8;    // B per-lane-half K offset

  v8f acc1 = {};            // sum_p om * tm
  v8f acc2 = {};            // sum_p sigmoid(om) * tm
#pragma unroll 4
  for (int k0 = 0; k0 < kP; k0 += 32) {
    v16h a  = pack_a(aRow + k0 + ka);
    v16h s  = pack_a(sRow + k0 + ka);
    v16h bb = pack_b(bRow + k0 + kb);
    acc1 = __builtin_amdgcn_wmma_f32_16x16x32_f16(false, a, false, bb,
                                                  (short)0, acc1, false, false);
    acc2 = __builtin_amdgcn_wmma_f32_16x16x32_f16(false, s, false, bb,
                                                  (short)0, acc2, false, false);
  }

  // C/D layout: VGPR r, lanes 0-15 -> (M=r, N=lane); lanes 16-31 -> (M=r+8).
  const int t = tBase + ll;
  const int lbl = (t < kT) ? labels[b * kT + t] : 0;
  const float tms = (t < kT) ? tmsum[b * kT + t] : 0.0f;
  const float invP = 1.0f / (float)kP;

#pragma unroll
  for (int r = 0; r < 8; ++r) {
    const int q = qBase + r + 8 * hi;
    if (q >= kQ || t >= kT) continue;
    const float S1 = acc1[r];
    const float S2 = acc2[r];
    const float cost_mask = (negsum[b * kQ + q] - S1) * invP;
    const float cost_dice =
        1.0f - (2.0f * S2 + 1.0f) / (sigsum[b * kQ + q] + tms + 1.0f);

    // class cost: softmax over 2 object logits, sigmoid class prob, sqrt mix
    const float* ol = pobj + (size_t)(b * kQ + q) * 2;
    const float mx = fmaxf(ol[0], ol[1]);
    const float e0 = expf(ol[0] - mx), e1 = expf(ol[1] - mx);
    const float inv = 1.0f / (e0 + e1);
    float prob;
    if (lbl == kC - 1) {
      prob = e1 * inv;
    } else {
      const float lg = plog[(size_t)(b * kQ + q) * kC + lbl];
      const float sg = 1.0f / (1.0f + expf(-lg));
      prob = sqrtf(sg * (e0 * inv));
    }
    out[((size_t)b * kQ + q) * kT + t] =
        5.0f * cost_mask + 2.0f * (-prob) + 5.0f * cost_dice;
  }
}

// ---------------------------------------------------------------------------
extern "C" void kernel_launch(void* const* d_in, const int* in_sizes, int n_in,
                              void* d_out, int out_size, void* d_ws, size_t ws_size,
                              hipStream_t stream) {
  (void)in_sizes; (void)n_in; (void)out_size; (void)ws_size;

  const float* pred_logits = (const float*)d_in[0];   // [B,Q,C]
  const float* pred_obj    = (const float*)d_in[1];   // [B,Q,2]
  const float* pred_masks  = (const float*)d_in[2];   // [B,Q,H,W]
  const float* tgt_masks   = (const float*)d_in[3];   // [B,T,H,W]
  const int*   tgt_labels  = (const int*)d_in[4];     // [B,T]
  const float* coords      = (const float*)d_in[5];   // [B,P,2]
  float* out = (float*)d_out;                         // [B,Q,T]

  // workspace carve-up (256B aligned)
  char* ws = (char*)d_ws;
  size_t off = 0;
  auto carve = [&](size_t bytes) -> char* {
    char* p = ws + off;
    off = (off + bytes + 255) & ~(size_t)255;
    return p;
  };
  _Float16* tm16 = (_Float16*)carve((size_t)kB * kT * kP * sizeof(_Float16));
  _Float16* om16 = (_Float16*)carve((size_t)kB * kQ * kP * sizeof(_Float16));
  _Float16* sg16 = (_Float16*)carve((size_t)kB * kQ * kP * sizeof(_Float16));
  float* partTm  = (float*)carve((size_t)kB * kT * kChunks * sizeof(float));
  float* partNeg = (float*)carve((size_t)kB * kQ * kChunks * sizeof(float));
  float* partSig = (float*)carve((size_t)kB * kQ * kChunks * sizeof(float));
  float* tmsum   = (float*)carve((size_t)kB * kT * sizeof(float));
  float* negsum  = (float*)carve((size_t)kB * kQ * sizeof(float));
  float* sigsum  = (float*)carve((size_t)kB * kQ * sizeof(float));

  // phase 1+2: bilinear sampling (HBM-bound), f16 packing, partial sums
  sample_tgt_kernel<<<dim3(kChunks, kT, kB), 256, 0, stream>>>(
      tgt_masks, coords, tm16, partTm);
  sample_pred_kernel<<<dim3(kChunks, kQ, kB), 256, 0, stream>>>(
      pred_masks, coords, om16, sg16, partNeg, partSig);

  // phase 3: deterministic chunk reductions (no float atomics)
  reduce_parts_kernel<<<(kB * kT + 255) / 256, 256, 0, stream>>>(partTm, tmsum, kB * kT);
  reduce_parts_kernel<<<(kB * kQ + 255) / 256, 256, 0, stream>>>(partNeg, negsum, kB * kQ);
  reduce_parts_kernel<<<(kB * kQ + 255) / 256, 256, 0, stream>>>(partSig, sigsum, kB * kQ);

  // phase 4: WMMA GEMMs over K=P + fused cost epilogue
  cost_matmul_kernel<<<dim3(kQTiles, kTTiles, kB), 32, 0, stream>>>(
      om16, sg16, tm16, negsum, sigsum, tmsum,
      pred_logits, pred_obj, tgt_labels, out);
}
